// GEN_74852690035376
// MI455X (gfx1250) — matrix operand order repaired and checked by reading, not compiled, not run on hardware
//
#include <hip/hip_runtime.h>
#include <hip/hip_bf16.h>

typedef _Float16 v16h __attribute__((ext_vector_type(16)));
typedef float    v8f  __attribute__((ext_vector_type(8)));
typedef unsigned int u32x4 __attribute__((ext_vector_type(4)));
typedef int          i32x8 __attribute__((ext_vector_type(8)));
typedef int          i32x4 __attribute__((ext_vector_type(4)));

#define TDIV(a,b) (((a)+(b)-1)/(b))
#define ASTR 34   // LDS tile row stride in halves: 32 data + 1-dword TDM pad

#if defined(__has_builtin)
#if __has_builtin(__builtin_amdgcn_tensor_load_to_lds) && __has_builtin(__builtin_amdgcn_s_wait_tensorcnt)
#define USE_TDM 1
#endif
#endif

// ---------------------------------------------------------------------------
// Utility kernels
// ---------------------------------------------------------------------------
__global__ void fill_kernel(float* __restrict__ p, float v, long n) {
  long i = (long)blockIdx.x * blockDim.x + threadIdx.x;
  if (i < n) p[i] = v;
}

// msg = relu(x[src]) + eps  (always > 0 -> uint-bit atomicMax with 0-init is exact)
__global__ void edge_max_kernel(const float* __restrict__ x, const int* __restrict__ ei,
                                float* __restrict__ m, int E, int d) {
  long i = (long)blockIdx.x * blockDim.x + threadIdx.x;
  if (i >= (long)E * d) return;
  int e = (int)(i / d), c = (int)(i % d);
  int s  = ei[e];
  int dn = ei[E + e];
  float msg = fmaxf(x[(long)s * d + c], 0.f) + 1e-7f;
  atomicMax((unsigned int*)(m + (long)dn * d + c), __float_as_uint(msg));
}

__global__ void edge_sum_kernel(const float* __restrict__ x, const int* __restrict__ ei,
                                const float* __restrict__ m, float* __restrict__ s,
                                float* __restrict__ t, int E, int d) {
  long i = (long)blockIdx.x * blockDim.x + threadIdx.x;
  if (i >= (long)E * d) return;
  int e = (int)(i / d), c = (int)(i % d);
  int sr = ei[e];
  int dn = ei[E + e];
  long off = (long)dn * d + c;
  float msg = fmaxf(x[(long)sr * d + c], 0.f) + 1e-7f;
  float ev = __expf(msg - m[off]);
  atomicAdd(s + off, ev);
  atomicAdd(t + off, msg * ev);
}

__global__ void agg_finalize_kernel(const float* __restrict__ t, const float* __restrict__ s,
                                    const float* __restrict__ x, float* __restrict__ hin, long n) {
  long i = (long)blockIdx.x * blockDim.x + threadIdx.x;
  if (i >= n) return;
  hin[i] = t[i] / (s[i] + 1e-16f) + x[i];
}

__global__ __launch_bounds__(256) void colstats_kernel(const float* __restrict__ h, int Nr, int C,
                                                       const float* __restrict__ gamma,
                                                       const float* __restrict__ beta,
                                                       float* __restrict__ scale,
                                                       float* __restrict__ shift) {
  int c = blockIdx.x;
  int tid = threadIdx.x;
  float s = 0.f, s2 = 0.f;
  for (int n = tid; n < Nr; n += 256) {
    float v = h[(long)n * C + c];
    s += v; s2 += v * v;
  }
  __shared__ float r1[256], r2[256];
  r1[tid] = s; r2[tid] = s2; __syncthreads();
  for (int w = 128; w > 0; w >>= 1) {
    if (tid < w) { r1[tid] += r1[tid + w]; r2[tid] += r2[tid + w]; }
    __syncthreads();
  }
  if (tid == 0) {
    float mean = r1[0] / (float)Nr;
    float var  = r2[0] / (float)Nr - mean * mean;
    float inv  = rsqrtf(var + 1e-5f);
    float sc   = gamma[c] * inv;
    scale[c] = sc;
    shift[c] = beta[c] - mean * sc;
  }
}

__global__ void bn_relu_kernel(float* __restrict__ h, const float* __restrict__ scale,
                               const float* __restrict__ shift, long n, int C) {
  long i = (long)blockIdx.x * blockDim.x + threadIdx.x;
  if (i >= n) return;
  int c = (int)(i % C);
  h[i] = fmaxf(h[i] * scale[c] + shift[c], 0.f);
}

__global__ void pool_cnt_kernel(const int* __restrict__ batch, float* __restrict__ cnt, int N) {
  int i = blockIdx.x * blockDim.x + threadIdx.x;
  if (i < N) atomicAdd(&cnt[batch[i]], 1.f);
}

__global__ void pool_sum_kernel(const float* __restrict__ h, const int* __restrict__ batch,
                                float* __restrict__ sums, int N, int C) {
  long i = (long)blockIdx.x * blockDim.x + threadIdx.x;
  if (i >= (long)N * C) return;
  int n = (int)(i / C), c = (int)(i % C);
  atomicAdd(&sums[(long)batch[n] * C + c], h[i]);
}

__global__ void pool_div_kernel(float* __restrict__ xg, const float* __restrict__ sums,
                                const float* __restrict__ cnt, int B, int C) {
  int i = blockIdx.x * blockDim.x + threadIdx.x;
  if (i >= B * C) return;
  xg[i] = sums[i] / fmaxf(cnt[i / C], 1.f);
}

// f32 [M,K] -> f16 [Mp,Kp], zero-padded
__global__ void cvt_a_f16_kernel(const float* __restrict__ A, _Float16* __restrict__ Ah,
                                 int M, int K, int Kp, long total) {
  long i = (long)blockIdx.x * blockDim.x + threadIdx.x;
  if (i >= total) return;
  int m = (int)(i / Kp), k = (int)(i % Kp);
  float v = (m < M && k < K) ? A[(long)m * K + k] : 0.f;
  Ah[i] = (_Float16)v;
}

// f32 [K,N] -> f16 B^T [Np,Kp], zero-padded
__global__ void cvt_bt_f16_kernel(const float* __restrict__ B, _Float16* __restrict__ Bt,
                                  int K, int N, int Kp, long total) {
  long i = (long)blockIdx.x * blockDim.x + threadIdx.x;
  if (i >= total) return;
  int n = (int)(i / Kp), k = (int)(i % Kp);
  float v = (n < N && k < K) ? B[(long)k * N + n] : 0.f;
  Bt[i] = (_Float16)v;
}

// ---------------------------------------------------------------------------
// TDM 2D tile load: f16 tile (tile_d0=32 elems, rows) from [*,Kp] tensor into
// LDS with 1-dword pad per 16 dwords (row stride ASTR=34 halves).
// D# per CDNA5 ISA. 6-arg builtin form.
// ---------------------------------------------------------------------------
#if defined(USE_TDM) && defined(__HIP_DEVICE_COMPILE__)
__device__ __forceinline__ void tdm_load_tile(unsigned lds_byte_off,
                                              const _Float16* gaddr,
                                              unsigned Kp, unsigned rows) {
  unsigned long long ga = (unsigned long long)(uintptr_t)gaddr;
  u32x4 g0;
  g0[0] = 1u;                                              // count=1, user
  g0[1] = lds_byte_off;                                    // lds_addr
  g0[2] = (unsigned)ga;                                    // global_addr[31:0]
  g0[3] = (unsigned)((ga >> 32) & 0x01FFFFFFu) | (2u << 30); // addr[56:32] | type=2
  i32x8 g1;
  g1[0] = (1 << 16) | (1 << 20) | (3 << 22);               // data=2B, pad_en, 16dw interval, 1dw pad
  g1[1] = (int)((Kp & 0xFFFFu) << 16);                     // tensor_dim0 lo16
  g1[2] = (int)((Kp >> 16) & 0xFFFFu) | (int)((rows & 0xFFFFu) << 16); // td0 hi | td1 lo
  g1[3] = (int)((rows >> 16) & 0xFFFFu) | (32 << 16);      // td1 hi | tile_dim0=32
  g1[4] = (int)(rows & 0xFFFFu);                           // tile_dim1=rows, tile_dim2=0
  g1[5] = (int)Kp;                                         // tensor_dim0_stride lo32
  g1[6] = 0;
  g1[7] = 0;
  i32x4 z4 = {0, 0, 0, 0};
  i32x8 z8 = {0, 0, 0, 0, 0, 0, 0, 0};
  __builtin_amdgcn_tensor_load_to_lds(g0, g1, z4, z4, z8, 0);
}
#endif

// ---------------------------------------------------------------------------
// GEMM: C[M,N] = op(A @ B + bias). A pre-converted f16 [Mp,Kp]; B pre-converted
// transposed f16 [Np,Kp]. Block tile 32(M) x 64(N), K-chunk 32, 8 waves (2x4),
// one v_wmma_f32_16x16x32_f16 per wave per chunk. Tiles staged via TDM
// (double-buffered) or branch-free vector loads.
// ---------------------------------------------------------------------------
__global__ __launch_bounds__(256) void gemm_f16wmma_kernel(
    const _Float16* __restrict__ Ah, const _Float16* __restrict__ Bt,
    const float* __restrict__ bias, float* __restrict__ C,
    int M, int N, int Kp, int ldc, int coff, int relu) {
  __shared__ _Float16 As[2][32 * ASTR];
  __shared__ _Float16 Bs[2][64 * ASTR];
  const int tid  = threadIdx.x;
  const int lane = tid & 31;
  const int wid  = tid >> 5;
  const int wm   = wid & 1;
  const int wn   = wid >> 1;
  const int m0   = blockIdx.y * 32;
  const int n0   = blockIdx.x * 64;
  const int lhalf = lane >> 4;
  const int l15   = lane & 15;
  const int nch   = Kp >> 5;
  v8f acc = {};

#if defined(USE_TDM) && defined(__HIP_DEVICE_COMPILE__)
  const unsigned ldsA[2] = { (unsigned)(uintptr_t)&As[0][0], (unsigned)(uintptr_t)&As[1][0] };
  const unsigned ldsB[2] = { (unsigned)(uintptr_t)&Bs[0][0], (unsigned)(uintptr_t)&Bs[1][0] };
  if (wid == 0) {
    tdm_load_tile(ldsA[0], Ah + (size_t)m0 * Kp, (unsigned)Kp, 32u);
    tdm_load_tile(ldsB[0], Bt + (size_t)n0 * Kp, (unsigned)Kp, 64u);
  }
#endif

  for (int ic = 0; ic < nch; ++ic) {
    const int cur = ic & 1;
    const int k0  = ic << 5;
#if defined(USE_TDM) && defined(__HIP_DEVICE_COMPILE__)
    if (wid == 0) {
      if (ic + 1 < nch) {
        const int nxt = cur ^ 1, k1 = k0 + 32;
        tdm_load_tile(ldsA[nxt], Ah + (size_t)m0 * Kp + k1, (unsigned)Kp, 32u);
        tdm_load_tile(ldsB[nxt], Bt + (size_t)n0 * Kp + k1, (unsigned)Kp, 64u);
        __builtin_amdgcn_s_wait_tensorcnt(2);   // current chunk's 2 loads done
      } else {
        __builtin_amdgcn_s_wait_tensorcnt(0);
      }
    }
#else
    {
      // branch-free staging from padded buffers (4 halves/thread A, 8 B)
      int mi = tid >> 3, kq = (tid & 7) * 4;
      const unsigned* ap = (const unsigned*)(Ah + (size_t)(m0 + mi) * Kp + k0 + kq);
      unsigned a0 = ap[0], a1 = ap[1];
      *(unsigned*)&As[cur][mi * ASTR + kq]     = a0;
      *(unsigned*)&As[cur][mi * ASTR + kq + 2] = a1;
      int ni = tid >> 2, kq2 = (tid & 3) * 8;
      const unsigned* bp = (const unsigned*)(Bt + (size_t)(n0 + ni) * Kp + k0 + kq2);
      unsigned b0 = bp[0], b1 = bp[1], b2 = bp[2], b3 = bp[3];
      *(unsigned*)&Bs[cur][ni * ASTR + kq2]     = b0;
      *(unsigned*)&Bs[cur][ni * ASTR + kq2 + 2] = b1;
      *(unsigned*)&Bs[cur][ni * ASTR + kq2 + 4] = b2;
      *(unsigned*)&Bs[cur][ni * ASTR + kq2 + 6] = b3;
    }
#endif
    __syncthreads();

    const _Float16* Ac = &As[cur][(wm * 16 + l15) * ASTR];
    const _Float16* Bc = &Bs[cur][(wn * 16 + l15) * ASTR];
    v16h af, bf;
    #pragma unroll
    for (int v = 0; v < 8; ++v) {
      int ka = ((v >> 2) << 4) + (lhalf << 3) + ((v & 3) << 1);  // A K-pair map (ISA 7.12.2)
      int kb = ((v >> 2) << 4) + (lhalf << 3) + ((v & 3) << 1);  // B K-pair map
      af[2 * v]     = Ac[ka];
      af[2 * v + 1] = Ac[ka + 1];
      bf[2 * v]     = Bc[kb];
      bf[2 * v + 1] = Bc[kb + 1];
    }
    acc = __builtin_amdgcn_wmma_f32_16x16x32_f16(false, af, false, bf,
                                                 (short)0, acc, false, false);
    __syncthreads();
  }

  const int colg  = n0 + wn * 16 + l15;
  const int mbase = m0 + wm * 16 + (lhalf << 3);
  if (colg < N) {
    float bv = bias ? bias[colg] : 0.f;
    #pragma unroll
    for (int r = 0; r < 8; ++r) {
      int rowg = mbase + r;
      if (rowg < M) {
        float v = acc[r] + bv;
        if (relu) v = fmaxf(v, 0.f);
        C[(long)rowg * ldc + colg + coff] = v;
      }
    }
  }
}

// ---------------------------------------------------------------------------
// Implicit-GEMM Conv1d: out[b,o,w] = sum_{i,t} emb[target[b,i], w+t]*cw[o,i,t]
// M = B*121 (rows (b,w), mult of 64), N = 32 channels, K = 8000 (i*8+t).
// B tiles (cwh f16 [32][8000], already B^T) via TDM; A gathered branch-free
// (8 contiguous f16 per thread, same protein index). Out as [b][o*121+w].
// ---------------------------------------------------------------------------
__global__ __launch_bounds__(256) void conv_wmma_kernel(
    const _Float16* __restrict__ embh,   // [26,128] f16
    const int*      __restrict__ target, // [B,1000]
    const _Float16* __restrict__ cwh,    // [32,8000] f16
    const float*    __restrict__ cb,     // [32]
    float* __restrict__ out, int M, int K) {
  __shared__ _Float16 As[2][64 * ASTR];
  __shared__ _Float16 Bs[2][32 * ASTR];
  const int tid  = threadIdx.x;
  const int lane = tid & 31;
  const int wid  = tid >> 5;
  const int wm   = wid >> 1;            // 4 wave-rows -> 64 M
  const int wn   = wid & 1;             // 2 wave-cols -> 32 N
  const int m0   = blockIdx.x * 64;
  const int lhalf = lane >> 4;
  const int l15   = lane & 15;
  const int nch   = K >> 5;
  v8f acc = {};

  // this thread's gather coordinates (fixed across chunks)
  const int mi  = tid >> 2;             // 0..63
  const int seg = tid & 3;              // which 8-block of the 32 k's
  const int gm  = m0 + mi;
  const int gb  = gm / 121, gw = gm % 121;
  const int tgt_base = gb * 1000;

#if defined(USE_TDM) && defined(__HIP_DEVICE_COMPILE__)
  const unsigned ldsB[2] = { (unsigned)(uintptr_t)&Bs[0][0], (unsigned)(uintptr_t)&Bs[1][0] };
  if (wid == 0) tdm_load_tile(ldsB[0], cwh, (unsigned)K, 32u);
#endif

  for (int ic = 0; ic < nch; ++ic) {
    const int cur = ic & 1;
    const int k0  = ic << 5;
    // A tile: 64x32 halves, 8 contiguous per thread (same i, t=0..7)
    {
      const int i   = (k0 >> 3) + seg;
      const int row = target[tgt_base + i];
      const _Float16* ep = embh + row * 128 + gw;
      _Float16 tv[8];
      #pragma unroll
      for (int j = 0; j < 8; ++j) tv[j] = ep[j];
      #pragma unroll
      for (int j = 0; j < 8; ++j) As[cur][mi * ASTR + seg * 8 + j] = tv[j];
    }
#if defined(USE_TDM) && defined(__HIP_DEVICE_COMPILE__)
    if (wid == 0) {
      if (ic + 1 < nch) {
        tdm_load_tile(ldsB[cur ^ 1], cwh + (k0 + 32), (unsigned)K, 32u);
        __builtin_amdgcn_s_wait_tensorcnt(1);
      } else {
        __builtin_amdgcn_s_wait_tensorcnt(0);
      }
    }
#else
    {
      int ni = tid >> 3, kq = (tid & 7) * 4;
      const unsigned* bp = (const unsigned*)(cwh + (size_t)ni * K + k0 + kq);
      unsigned b0 = bp[0], b1 = bp[1];
      *(unsigned*)&Bs[cur][ni * ASTR + kq]     = b0;
      *(unsigned*)&Bs[cur][ni * ASTR + kq + 2] = b1;
    }
#endif
    __syncthreads();

    const _Float16* Ac = &As[cur][(wm * 16 + l15) * ASTR];
    const _Float16* Bc = &Bs[cur][(wn * 16 + l15) * ASTR];
    v16h af, bf;
    #pragma unroll
    for (int v = 0; v < 8; ++v) {
      int ka = ((v >> 2) << 4) + (lhalf << 3) + ((v & 3) << 1);
      int kb = ((v >> 2) << 4) + (lhalf << 3) + ((v & 3) << 1);
      af[2 * v]     = Ac[ka];
      af[2 * v + 1] = Ac[ka + 1];
      bf[2 * v]     = Bc[kb];
      bf[2 * v + 1] = Bc[kb + 1];
    }
    acc = __builtin_amdgcn_wmma_f32_16x16x32_f16(false, af, false, bf,
                                                 (short)0, acc, false, false);
    __syncthreads();
  }

  const int colg  = wn * 16 + l15;                  // channel o
  const int mbase = m0 + wm * 16 + (lhalf << 3);
  float bv = cb[colg];
  #pragma unroll
  for (int r = 0; r < 8; ++r) {
    int g = mbase + r;
    if (g < M) {
      int b = g / 121, w = g % 121;
      out[(long)b * 3872 + colg * 121 + w] = acc[r] + bv;
    }
  }
}

// ---------------------------------------------------------------------------
// Host orchestration
// ---------------------------------------------------------------------------
extern "C" void kernel_launch(void* const* d_in, const int* in_sizes, int n_in,
                              void* d_out, int out_size, void* d_ws, size_t ws_size,
                              hipStream_t stream) {
  (void)n_in; (void)out_size; (void)ws_size;
  const float* x0     = (const float*)d_in[0];
  const int*   ei     = (const int*)d_in[1];
  const int*   batch  = (const int*)d_in[2];
  const int*   target = (const int*)d_in[3];
  const int N = in_sizes[0] / 78;
  const int E = in_sizes[1] / 2;
  const int B = in_sizes[3] / 1000;

  const float *gw1[3], *gb1[3], *gg[3], *gbe[3], *gw2[3], *gb2[3];
  for (int l = 0; l < 3; ++l) {
    gw1[l] = (const float*)d_in[4 + 6 * l + 0];
    gb1[l] = (const float*)d_in[4 + 6 * l + 1];
    gg[l]  = (const float*)d_in[4 + 6 * l + 2];
    gbe[l] = (const float*)d_in[4 + 6 * l + 3];
    gw2[l] = (const float*)d_in[4 + 6 * l + 4];
    gb2[l] = (const float*)d_in[4 + 6 * l + 5];
  }
  const float* fg1_w = (const float*)d_in[22]; const float* fg1_b = (const float*)d_in[23];
  const float* fg2_w = (const float*)d_in[24]; const float* fg2_b = (const float*)d_in[25];
  const float* fg3_w = (const float*)d_in[26]; const float* fg3_b = (const float*)d_in[27];
  const float* emb   = (const float*)d_in[28];
  const float* cw    = (const float*)d_in[29]; const float* cb    = (const float*)d_in[30];
  const float* fxt_w = (const float*)d_in[31]; const float* fxt_b = (const float*)d_in[32];
  const float* f1_w  = (const float*)d_in[33]; const float* f1_b  = (const float*)d_in[34];
  const float* f2_w  = (const float*)d_in[35]; const float* f2_b  = (const float*)d_in[36];
  const float* o_w   = (const float*)d_in[37]; const float* o_b   = (const float*)d_in[38];

  auto r32 = [](int v) { return (v + 31) & ~31; };
  auto r64 = [](int v) { return (v + 63) & ~63; };

  // workspace allocator (bytes)
  char* base = (char*)d_ws;
  size_t off = 0;
  auto allocf = [&](size_t n) { float* p = (float*)(base + off); off += n * 4; return p; };
  auto alloch = [&](size_t n) { off = (off + 3) & ~(size_t)3;
                                _Float16* p = (_Float16*)(base + off); off += n * 2; return p; };

  float* featA = allocf((size_t)N * 312);
  float* featB = allocf((size_t)N * 624);
  float* mbuf  = allocf((size_t)N * 312);
  float* sbuf  = allocf((size_t)N * 312);
  float* tbuf  = allocf((size_t)N * 312);
  float* hmid  = allocf((size_t)N * 624);
  float* scale = allocf(1024);
  float* shift = allocf(1024);
  float* sums  = allocf((size_t)B * 624);
  float* cnt   = allocf(B);
  float* xg    = allocf((size_t)B * 624);
  float* t1    = allocf((size_t)B * 1024);
  float* t2    = allocf((size_t)B * 1024);
  float* convo = allocf((size_t)B * 3872);
  float* xc    = allocf((size_t)B * 256);
  _Float16* AhBuf = alloch((size_t)r32(N) * 640);   // shared f16 A scratch

  // convert a weight [K,N] to f16 B^T [Np,Kp]
  auto cvtw = [&](const float* W, int K, int Nc) {
    int Kp = r32(K), Np = r64(Nc);
    _Float16* p = alloch((size_t)Np * Kp);
    long tot = (long)Np * Kp;
    cvt_bt_f16_kernel<<<TDIV(tot, 256), 256, 0, stream>>>(W, p, K, Nc, Kp, tot);
    return p;
  };
  _Float16* w1t[3], * w2t[3];
  int din_[3] = {78, 78, 312}, dhid_[3] = {156, 156, 624}, dout_[3] = {78, 312, 624};
  for (int l = 0; l < 3; ++l) {
    w1t[l] = cvtw(gw1[l], din_[l], dhid_[l]);
    w2t[l] = cvtw(gw2[l], dhid_[l], dout_[l]);
  }
  _Float16* fg1t = cvtw(fg1_w, 624, 512);
  _Float16* fg2t = cvtw(fg2_w, 512, 1024);
  _Float16* fg3t = cvtw(fg3_w, 1024, 128);
  _Float16* fxtt = cvtw(fxt_w, 3872, 128);
  _Float16* f1t  = cvtw(f1_w, 256, 1024);
  _Float16* f2t  = cvtw(f2_w, 1024, 512);
  _Float16* ot   = cvtw(o_w, 512, 1);
  _Float16* embh = alloch((size_t)26 * 128);
  _Float16* cwh  = alloch((size_t)32 * 8000);
  cvt_a_f16_kernel<<<TDIV(26 * 128, 256), 256, 0, stream>>>(emb, embh, 26, 128, 128, 26 * 128);
  cvt_a_f16_kernel<<<TDIV(32 * 8000, 256), 256, 0, stream>>>(cw, cwh, 32, 8000, 8000, 32 * 8000);

  // GEMM from f32 A: convert to padded f16, then WMMA kernel
  auto gemmf = [&](const float* A, int M, int K, const _Float16* Bth, int Nc,
                   const float* bias, float* C, int ldc, int coff, int relu) {
    int Kp = r32(K), Mp = r32(M);
    long tot = (long)Mp * Kp;
    cvt_a_f16_kernel<<<TDIV(tot, 256), 256, 0, stream>>>(A, AhBuf, M, K, Kp, tot);
    dim3 grid(TDIV(Nc, 64), TDIV(M, 32));
    gemm_f16wmma_kernel<<<grid, 256, 0, stream>>>(AhBuf, Bth, bias, C, M, Nc, Kp, ldc, coff, relu);
  };

  auto run_layer = [&](const float* xin, int d_in_, int l, int d_hid, int d_out2, float* out) {
    long nd = (long)N * d_in_;
    fill_kernel<<<TDIV(nd, 256), 256, 0, stream>>>(mbuf, 0.f, nd);
    fill_kernel<<<TDIV(nd, 256), 256, 0, stream>>>(sbuf, 0.f, nd);
    fill_kernel<<<TDIV(nd, 256), 256, 0, stream>>>(tbuf, 0.f, nd);
    long ed = (long)E * d_in_;
    edge_max_kernel<<<TDIV(ed, 256), 256, 0, stream>>>(xin, ei, mbuf, E, d_in_);
    edge_sum_kernel<<<TDIV(ed, 256), 256, 0, stream>>>(xin, ei, mbuf, sbuf, tbuf, E, d_in_);
    agg_finalize_kernel<<<TDIV(nd, 256), 256, 0, stream>>>(tbuf, sbuf, xin, tbuf, nd);
    gemmf(tbuf, N, d_in_, w1t[l], d_hid, gb1[l], hmid, d_hid, 0, 0);
    colstats_kernel<<<d_hid, 256, 0, stream>>>(hmid, N, d_hid, gg[l], gbe[l], scale, shift);
    long nh = (long)N * d_hid;
    bn_relu_kernel<<<TDIV(nh, 256), 256, 0, stream>>>(hmid, scale, shift, nh, d_hid);
    gemmf(hmid, N, d_hid, w2t[l], d_out2, gb2[l], out, d_out2, 0, 1);
  };

  run_layer(x0,    78,  0, 156, 78,  featB);
  run_layer(featB, 78,  1, 156, 312, featA);
  run_layer(featA, 312, 2, 624, 624, featB);

  // global mean pool
  long bc = (long)B * 624;
  fill_kernel<<<TDIV(bc, 256), 256, 0, stream>>>(sums, 0.f, bc);
  fill_kernel<<<TDIV(B, 256), 256, 0, stream>>>(cnt, 0.f, B);
  pool_cnt_kernel<<<TDIV(N, 256), 256, 0, stream>>>(batch, cnt, N);
  long nc = (long)N * 624;
  pool_sum_kernel<<<TDIV(nc, 256), 256, 0, stream>>>(featB, batch, sums, N, 624);
  pool_div_kernel<<<TDIV(bc, 256), 256, 0, stream>>>(xg, sums, cnt, B, 624);

  // graph FC head -> xc[:, :128]
  gemmf(xg, B, 624, fg1t, 512, fg1_b, t1, 512, 0, 1);
  gemmf(t1, B, 512, fg2t, 1024, fg2_b, t2, 1024, 0, 1);
  gemmf(t2, B, 1024, fg3t, 128, fg3_b, xc, 256, 0, 1);

  // protein branch -> xc[:, 128:]
  int Mc = B * 121;
  conv_wmma_kernel<<<TDIV(Mc, 64), 256, 0, stream>>>(embh, target, cwh, cb, convo, Mc, 8000);
  gemmf(convo, B, 3872, fxtt, 128, fxt_b, xc, 256, 128, 0);

  // combined head
  gemmf(xc, B, 256, f1t, 1024, f1_b, t1, 1024, 0, 1);
  gemmf(t1, B, 1024, f2t, 512, f2_b, t2, 512, 0, 1);
  gemmf(t2, B, 512, ot, 1, o_b, (float*)d_out, 1, 0, 0);
}
